// GraphAttentionLayer_61795989455255
// MI455X (gfx1250) — matrix operand order repaired and checked by reading, not compiled, not run on hardware
//
#include <hip/hip_runtime.h>

// ---------------------------------------------------------------------------
// GAT layer for MI455X (gfx1250, wave32, WMMA).
// N=8192, IN_F=256, OUT_F=128, ALPHA=0.2, MASK_VAL=-9e15, CLAMP=1e10
// ---------------------------------------------------------------------------

#define N_NODES 8192
#define IN_F    256
#define OUT_F   128

typedef __attribute__((ext_vector_type(16))) __bf16 v16bf;
typedef __attribute__((ext_vector_type(8)))  float  v8f;

// Native f32 -> bf16 (backend emits v_cvt_pk_bf16_f32-class ops, 2 per instr)
static __device__ __forceinline__ __bf16 f2bf(float f) {
  return static_cast<__bf16>(f);
}

static __device__ __forceinline__ float score_fn(float ff1, float f2v) {
  float s = ff1 + f2v;
  s = s > 0.f ? s : 0.2f * s;                        // leaky relu
  return fminf(fmaxf(s, -1e10f), 1e10f);             // clamp
}

// ---------------------------------------------------------------------------
// K0: WbT[f][k] = bf16(W[k][f])   (128 x 256 bf16, feature-major)
// ---------------------------------------------------------------------------
__global__ void k_wbt(const float* __restrict__ W, __bf16* __restrict__ wbT) {
  int idx = blockIdx.x * blockDim.x + threadIdx.x;   // 0..32767
  if (idx >= OUT_F * IN_F) return;
  int f = idx >> 8;          // /256
  int k = idx & 255;
  wbT[idx] = f2bf(W[k * OUT_F + f]);
}

// ---------------------------------------------------------------------------
// K1: Wh = x @ W  via v_wmma_f32_16x16x32_bf16 (A from x fp32 converted
// on the fly, B from WbT). Writes Wh fp32 and WhT bf16 (feature-major).
// One wave per 16x16 output tile; 512*8 = 4096 waves.
// ---------------------------------------------------------------------------
__global__ __launch_bounds__(128)
void k_gemm_wh(const float* __restrict__ x, const __bf16* __restrict__ wbT,
               float* __restrict__ wh, __bf16* __restrict__ whT) {
  int wid  = blockIdx.x * (blockDim.x >> 5) + (threadIdx.x >> 5);
  int lane = threadIdx.x & 31;
  int mtile = wid >> 3;                  // 0..511
  int ntile = wid & 7;                   // 0..7
  int i0 = mtile * 16, n0 = ntile * 16;
  int half = lane >> 4;                  // lane group
  int l16  = lane & 15;
  int kb1  = half ? 16 : 0;              // A K-run offsets for this lane group
  const float*  xrow = x + (size_t)(i0 + l16) * IN_F;
  const __bf16* bcol = wbT + (size_t)(n0 + l16) * IN_F + (half ? 16 : 0);

  v8f c = {};
#pragma unroll
  for (int k0 = 0; k0 < IN_F; k0 += 32) {
    v16bf a;
#pragma unroll
    for (int ch = 0; ch < 4; ++ch) {
      int koff = k0 + kb1 + ch * 4;
      float4 q = *(const float4*)(xrow + koff);
      a[ch * 4 + 0] = f2bf(q.x);
      a[ch * 4 + 1] = f2bf(q.y);
      a[ch * 4 + 2] = f2bf(q.z);
      a[ch * 4 + 3] = f2bf(q.w);
    }
    v16bf b = *(const v16bf*)(bcol + k0);
    c = __builtin_amdgcn_wmma_f32_16x16x32_bf16(false, a, false, b,
                                                (short)0, c, false, false);
  }
  int mb = i0 + (half ? 8 : 0);
#pragma unroll
  for (int r = 0; r < 8; ++r) {
    int row = mb + r;
    float v = c[r];
    wh [(size_t)row * OUT_F + n0 + l16]       = v;
    whT[(size_t)(n0 + l16) * N_NODES + row]   = f2bf(v);
  }
}

// ---------------------------------------------------------------------------
// K2: f1[i] = Wh[i,:] . a[:128],  f2[i] = Wh[i,:] . a[128:]
// ---------------------------------------------------------------------------
__global__ void k_f12(const float* __restrict__ wh, const float* __restrict__ a,
                      float* __restrict__ f1, float* __restrict__ f2) {
  int i = blockIdx.x * blockDim.x + threadIdx.x;
  if (i >= N_NODES) return;
  const float* r = wh + (size_t)i * OUT_F;
  float s1 = 0.f, s2 = 0.f;
#pragma unroll 4
  for (int k = 0; k < OUT_F; k += 4) {
    float4 v  = *(const float4*)(r + k);
    float4 a1 = *(const float4*)(a + k);
    float4 a2 = *(const float4*)(a + OUT_F + k);
    s1 += v.x * a1.x + v.y * a1.y + v.z * a1.z + v.w * a1.w;
    s2 += v.x * a2.x + v.y * a2.y + v.z * a2.z + v.w * a2.w;
  }
  f1[i] = s1; f2[i] = s2;
}

// ---------------------------------------------------------------------------
// K3: per-row online softmax stats over masked scores.
// One wave per row; lane-local online (max, sum), then shuffle-combine.
// ---------------------------------------------------------------------------
__global__ __launch_bounds__(256)
void k_rowstats(const float* __restrict__ adj, const float* __restrict__ f1,
                const float* __restrict__ f2, float* __restrict__ rowmax,
                float* __restrict__ invden) {
  int row  = blockIdx.x * (blockDim.x >> 5) + (threadIdx.x >> 5);
  int lane = threadIdx.x & 31;
  if (row >= N_NODES) return;
  const float* ap = adj + (size_t)row * N_NODES;
  float ff1 = f1[row];
  float m = -3.0e38f, s = 0.f;

  auto upd = [&](float adjv, float f2v) {
    if (adjv > 0.f) {
      float sc = score_fn(ff1, f2v);
      if (sc > m) { s *= __expf(m - sc); m = sc; }
      s += __expf(sc - m);
    }
  };
  for (int it = 0; it < N_NODES / (32 * 4); ++it) {
    int j = (it * 32 + lane) * 4;
    float4 av = *(const float4*)(ap + j);
    float4 fv = *(const float4*)(f2 + j);
    upd(av.x, fv.x); upd(av.y, fv.y); upd(av.z, fv.z); upd(av.w, fv.w);
  }
  // wave32 tree combine of (m, s)
#pragma unroll
  for (int off = 16; off > 0; off >>= 1) {
    float mo = __shfl_xor(m, off, 32);
    float so = __shfl_xor(s, off, 32);
    float mn = fmaxf(m, mo);
    s = s * __expf(m - mn) + so * __expf(mo - mn);
    m = mn;
  }
  if (lane == 0) { rowmax[row] = m; invden[row] = 1.f / s; }
}

// ---------------------------------------------------------------------------
// K4: fused attention @ Wh + ELU.
// Workgroup = 4 waves, one 16-row block, all 128 features.
// Waves split the 8192-column K dimension; per 16x32 tile each wave computes
// normalized probabilities p (fp32 -> bf16 A-tile) and does 8 WMMAs against
// WhT bf16 B-tiles. Cross-wave reduce in LDS, ELU, store.
// ---------------------------------------------------------------------------
#define LPAD 132   // LDS row stride (floats) to dodge bank conflicts

__global__ __launch_bounds__(128)
void k_attn(const float* __restrict__ adj, const __bf16* __restrict__ whT,
            const float* __restrict__ f1, const float* __restrict__ f2,
            const float* __restrict__ rowmax, const float* __restrict__ invden,
            float* __restrict__ out) {
  __shared__ float lds[4 * 16 * LPAD];
  int wave = threadIdx.x >> 5;
  int lane = threadIdx.x & 31;
  int half = lane >> 4;
  int l16  = lane & 15;
  int i0   = blockIdx.x * 16;
  int row  = i0 + l16;                    // A-tile row owned by this lane
  float ff1 = f1[row];
  float fmx = rowmax[row];
  float fid = invden[row];
  int kb1 = half ? 16 : 0;                // A K-run offsets (K=32 tile)
  int kb2 = kb1 + 8;
  const float* arow = adj + (size_t)row * N_NODES;

  v8f c[8];
  v8f z = {};
#pragma unroll
  for (int fch = 0; fch < 8; ++fch) c[fch] = z;

  for (int t = 0; t < N_NODES / (32 * 4); ++t) {   // 64 tiles per wave
    int j0 = (t * 4 + wave) * 32;
    // prefetch next adj tile for this wave (global_prefetch_b8)
    if (t + 1 < N_NODES / (32 * 4))
      __builtin_prefetch(arow + j0 + 128 + kb1, 0, 1);

    v16bf a;
#pragma unroll
    for (int ch = 0; ch < 4; ++ch) {
      int koff = (ch < 2 ? kb1 + ch * 4 : kb2 + (ch - 2) * 4);
      float4 q = *(const float4*)(arow + j0 + koff);
      float4 g = *(const float4*)(f2 + j0 + koff);
      float p0 = q.x > 0.f ? __expf(score_fn(ff1, g.x) - fmx) * fid : 0.f;
      float p1 = q.y > 0.f ? __expf(score_fn(ff1, g.y) - fmx) * fid : 0.f;
      float p2 = q.z > 0.f ? __expf(score_fn(ff1, g.z) - fmx) * fid : 0.f;
      float p3 = q.w > 0.f ? __expf(score_fn(ff1, g.w) - fmx) * fid : 0.f;
      a[ch * 4 + 0] = f2bf(p0);
      a[ch * 4 + 1] = f2bf(p1);
      a[ch * 4 + 2] = f2bf(p2);
      a[ch * 4 + 3] = f2bf(p3);
    }
#pragma unroll
    for (int fch = 0; fch < 8; ++fch) {
      v16bf b = *(const v16bf*)(whT + (size_t)(fch * 16 + l16) * N_NODES +
                                j0 + (half ? 16 : 0));
      c[fch] = __builtin_amdgcn_wmma_f32_16x16x32_bf16(false, a, false, b,
                                                       (short)0, c[fch],
                                                       false, false);
    }
  }

  // spill per-wave partials to LDS (C layout: row = r + half*8, col = l16)
  int mb = half ? 8 : 0;
#pragma unroll
  for (int fch = 0; fch < 8; ++fch)
#pragma unroll
    for (int r = 0; r < 8; ++r)
      lds[(wave * 16 + mb + r) * LPAD + fch * 16 + l16] = c[fch][r];
  __syncthreads();

  // thread t owns feature column t across all 16 rows
  int col = threadIdx.x;                  // 0..127
#pragma unroll
  for (int r2 = 0; r2 < 16; ++r2) {
    float v = lds[(0 * 16 + r2) * LPAD + col] + lds[(1 * 16 + r2) * LPAD + col]
            + lds[(2 * 16 + r2) * LPAD + col] + lds[(3 * 16 + r2) * LPAD + col];
    v = v > 0.f ? v : __expf(v) - 1.f;    // ELU
    out[(size_t)(i0 + r2) * OUT_F + col] = v;
  }
}

// ---------------------------------------------------------------------------
// Host launcher
// ---------------------------------------------------------------------------
extern "C" void kernel_launch(void* const* d_in, const int* in_sizes, int n_in,
                              void* d_out, int out_size, void* d_ws, size_t ws_size,
                              hipStream_t stream) {
  (void)in_sizes; (void)n_in; (void)out_size; (void)ws_size;
  const float* x   = (const float*)d_in[0];   // [8192,256]
  const float* adj = (const float*)d_in[1];   // [8192,8192]
  const float* W   = (const float*)d_in[2];   // [256,128]
  const float* a   = (const float*)d_in[3];   // [256,1]
  float* out = (float*)d_out;                 // [8192,128]

  char* ws = (char*)d_ws;
  float*  wh   = (float*)ws;                                  // 4 MB fp32
  __bf16* whT  = (__bf16*)(ws + (4u << 20));                  // 2 MB bf16 [128][8192]
  __bf16* wbT  = (__bf16*)(ws + (6u << 20));                  // 64 KB bf16 [128][256]
  float*  f1   = (float*)(ws + (6u << 20) + (64u << 10));     // 32 KB
  float*  f2   = f1 + N_NODES;
  float*  rmx  = f2 + N_NODES;
  float*  idn  = rmx + N_NODES;

  k_wbt     <<<128,  256, 0, stream>>>(W, wbT);
  k_gemm_wh <<<1024, 128, 0, stream>>>(x, wbT, wh, whT);
  k_f12     <<<32,   256, 0, stream>>>(wh, a, f1, f2);
  k_rowstats<<<1024, 256, 0, stream>>>(adj, f1, f2, rmx, idn);
  k_attn    <<<512,  128, 0, stream>>>(adj, whT, f1, f2, rmx, idn, out);
}
